// DAEGC_68161130987969
// MI455X (gfx1250) — matrix operand order repaired and checked by reading, not compiled, not run on hardware
//
#include <hip/hip_runtime.h>

// ---------------------------------------------------------------------------
// MI455X (gfx1250, wave32) implementation.
//
// mass^2[n,c] = sum_f T[f][c][ s_{n,f}+1 ]  with T a precomputed
// [16 features][16 clusters][102 bins] table that absorbs the cumsum /
// min/max / is_zero logic.  Expressing the per-sample bin index as a one-hot
// over the 102-wide axis turns the per-16-sample-tile computation into
//   D[16x16] += Onehot_f[16x32] x T_f[32x16]   (4 K-chunks per feature)
// which maps exactly onto v_wmma_f32_16x16x32_f16 with f32 accumulation.
// Everything is HBM-trivial (~20 MB total traffic), so the design goal is a
// single streaming pass over z with the 16x16 inner product on the WMMA path.
// ---------------------------------------------------------------------------

typedef __attribute__((ext_vector_type(16))) _Float16 v16h;
typedef __attribute__((ext_vector_type(8)))  float    v8f;

#define N_SAMPLES 100000
#define F_DIM     16
#define C_CLUST   16
#define NBINS     100
#define NEDGE     101
#define COMB      (N_SAMPLES + C_CLUST)
#define A_RANGE   102                  // bin index s+1 in [0, 101]
#define A_PAD     112                  // pad so [f][c][a] rows are 224B = 7*32B
#define N_TILES   (N_SAMPLES / 16)     // 6250 exact
#define DIVISOR   100016.0f

// workspace layout (bytes)
#define WS_LO   0                      // 16 f32
#define WS_HI   64                     // 16 f32
#define WS_CNT  128                    // 16*100 u32
#define WS_EDG  6528                   // 16*101 f32
#define WS_T2   13056                  // 16*16*112 f16, layout [f][c][a]
#define T2_HALFS (F_DIM * C_CLUST * A_PAD)   // 28672

// -------------------------- helpers ---------------------------------------

__device__ __forceinline__ void atomicMinF(float* a, float v) {
    if (v >= 0.0f) atomicMin((int*)a, __float_as_int(v));
    else           atomicMax((unsigned int*)a, __float_as_uint(v));
}
__device__ __forceinline__ void atomicMaxF(float* a, float v) {
    if (v >= 0.0f) atomicMax((int*)a, __float_as_int(v));
    else           atomicMin((unsigned int*)a, __float_as_uint(v));
}

// searchsorted(edges, v, 'left'): #{j : e_j < v}.  Arithmetic guess + fixup.
__device__ __forceinline__ int ssorted_left(const float* E, float v) {
    float lo = E[0], hi = E[NEDGE - 1];
    float invw = (hi > lo) ? (float)NBINS / (hi - lo) : 0.0f;
    int p = (int)fminf(fmaxf((v - lo) * invw, 0.0f), (float)NEDGE);
    while (p < NEDGE && E[p] < v)      ++p;
    while (p > 0     && E[p - 1] >= v) --p;
    return p;
}
// searchsorted(edges, v, 'right'): #{j : e_j <= v}
__device__ __forceinline__ int ssorted_right(const float* E, float v) {
    float lo = E[0], hi = E[NEDGE - 1];
    float invw = (hi > lo) ? (float)NBINS / (hi - lo) : 0.0f;
    int p = (int)fminf(fmaxf((v - lo) * invw, 0.0f), (float)NEDGE);
    while (p < NEDGE && E[p] <= v)    ++p;
    while (p > 0     && E[p - 1] > v) --p;
    return p;
}

// -------------------------- phase kernels ----------------------------------

__global__ void k_init(float* lo, float* hi, unsigned* cnt) {
    int t = threadIdx.x;
    if (t < F_DIM) { lo[t] = __builtin_inff(); hi[t] = -__builtin_inff(); }
    for (int i = t; i < F_DIM * NBINS; i += blockDim.x) cnt[i] = 0u;
}

__global__ void k_minmax(const float* __restrict__ z, const float* __restrict__ cl,
                         float* lo, float* hi) {
    float lmin[F_DIM], lmax[F_DIM];
#pragma unroll
    for (int f = 0; f < F_DIM; ++f) { lmin[f] = __builtin_inff(); lmax[f] = -__builtin_inff(); }
    for (int row = blockIdx.x * blockDim.x + threadIdx.x; row < COMB;
         row += gridDim.x * blockDim.x) {
        const float4* r = (const float4*)((row < N_SAMPLES) ? (z + (size_t)row * F_DIM)
                                                            : (cl + (size_t)(row - N_SAMPLES) * F_DIM));
        float4 a = r[0], b = r[1], c = r[2], d = r[3];
        float v[F_DIM] = {a.x,a.y,a.z,a.w, b.x,b.y,b.z,b.w, c.x,c.y,c.z,c.w, d.x,d.y,d.z,d.w};
#pragma unroll
        for (int f = 0; f < F_DIM; ++f) {
            lmin[f] = fminf(lmin[f], v[f]);
            lmax[f] = fmaxf(lmax[f], v[f]);
        }
    }
#pragma unroll
    for (int f = 0; f < F_DIM; ++f) {
#pragma unroll
        for (int off = 16; off >= 1; off >>= 1) {
            lmin[f] = fminf(lmin[f], __shfl_xor(lmin[f], off));
            lmax[f] = fmaxf(lmax[f], __shfl_xor(lmax[f], off));
        }
    }
    if ((threadIdx.x & 31) == 0) {
#pragma unroll
        for (int f = 0; f < F_DIM; ++f) { atomicMinF(&lo[f], lmin[f]); atomicMaxF(&hi[f], lmax[f]); }
    }
}

__global__ void k_edges(const float* lo, const float* hi, float* edg) {
    for (int i = threadIdx.x; i < F_DIM * NEDGE; i += blockDim.x) {
        int f = i / NEDGE, j = i % NEDGE;
        float t = (float)j / (float)NBINS;
        edg[i] = lo[f] + (hi[f] - lo[f]) * t;
    }
}

__global__ void k_hist(const float* __restrict__ z, const float* __restrict__ cl,
                       const float* __restrict__ edg, unsigned* cnt) {
    __shared__ unsigned hc[F_DIM * NBINS];
    __shared__ float E[F_DIM * NEDGE];
    for (int i = threadIdx.x; i < F_DIM * NBINS; i += blockDim.x) hc[i] = 0u;
    for (int i = threadIdx.x; i < F_DIM * NEDGE; i += blockDim.x) E[i] = edg[i];
    __syncthreads();
    for (int row = blockIdx.x * blockDim.x + threadIdx.x; row < COMB;
         row += gridDim.x * blockDim.x) {
        const float4* r = (const float4*)((row < N_SAMPLES) ? (z + (size_t)row * F_DIM)
                                                            : (cl + (size_t)(row - N_SAMPLES) * F_DIM));
        float4 a = r[0], b = r[1], c = r[2], d = r[3];
        float v[F_DIM] = {a.x,a.y,a.z,a.w, b.x,b.y,b.z,b.w, c.x,c.y,c.z,c.w, d.x,d.y,d.z,d.w};
#pragma unroll
        for (int f = 0; f < F_DIM; ++f) {
            int p = ssorted_right(&E[f * NEDGE], v[f]) - 1;      // right-closed last bin
            p = min(max(p, 0), NBINS - 1);
            atomicAdd(&hc[f * NBINS + p], 1u);
        }
    }
    __syncthreads();
    for (int i = threadIdx.x; i < F_DIM * NBINS; i += blockDim.x)
        if (hc[i]) atomicAdd(&cnt[i], hc[i]);
}

// cumsum + cluster bins + fold everything into T2[f][c][a] = (sum_counts/div)^2
__global__ void k_table(const float* __restrict__ cl, const float* __restrict__ edg,
                        const unsigned* __restrict__ cnt, _Float16* __restrict__ t2) {
    __shared__ float cum[F_DIM * NBINS];
    __shared__ int   cb[C_CLUST * F_DIM];
    int t = threadIdx.x;
    if (t < F_DIM) {
        float s = 0.0f;
        for (int j = 0; j < NBINS; ++j) { s += (float)cnt[t * NBINS + j]; cum[t * NBINS + j] = s; }
    }
    if (t < C_CLUST * F_DIM) {
        int c = t >> 4, f = t & 15;
        cb[c * F_DIM + f] = ssorted_left(&edg[f * NEDGE], cl[c * F_DIM + f]) - 1;  // in [-1,100]
    }
    __syncthreads();
    const float inv = 1.0f / DIVISOR;
    for (int i = t; i < F_DIM * C_CLUST * A_PAD; i += blockDim.x) {
        int f = i / (C_CLUST * A_PAD);
        int rem = i % (C_CLUST * A_PAD);
        int c = rem / A_PAD;
        int a = rem % A_PAD;
        float val = 0.0f;
        if (a < A_RANGE) {
            int s = a - 1;                      // sample bin
            int k = cb[c * F_DIM + f];          // cluster bin
            int mn = min(s, k), mx = max(s, k);
            float hiS = cum[f * NBINS + min(max(mx, 0), NBINS - 1)];
            float sc  = (mn <= 0) ? hiS : (hiS - cum[f * NBINS + (mn - 1)]);
            val = sc * inv;
            val = val * val;
        }
        t2[i] = (_Float16)val;
    }
}

// -------------------------- main WMMA kernel -------------------------------

#define WAVES_PER_BLOCK 8

__global__ __launch_bounds__(256, 1)
void k_mass_wmma(const float* __restrict__ z, const float* __restrict__ edg,
                 const _Float16* __restrict__ t2, float* __restrict__ out) {
    // LDS: T2 table [f][c][A_PAD] (57.4 KB) + edges (6.5 KB)  < 64 KB static.
    __shared__ __align__(32) _Float16 sT2[T2_HALFS + 32];   // +32 pad: safe B over-read
    __shared__ float E[F_DIM * NEDGE];

    for (int i = threadIdx.x; i < T2_HALFS / 8; i += blockDim.x)
        ((uint4*)sT2)[i] = ((const uint4*)t2)[i];           // 16B vector copy
    for (int i = threadIdx.x; i < 32; i += blockDim.x) sT2[T2_HALFS + i] = (_Float16)0.0f;
    for (int i = threadIdx.x; i < F_DIM * NEDGE; i += blockDim.x) E[i] = edg[i];
    __syncthreads();

    const int lane = threadIdx.x & 31;
    const int wid  = threadIdx.x >> 5;
    const int hi   = lane >> 4;        // half-wave: K halves of A, row offset of C/D
    const int ncol = lane & 15;        // B/D column, and A row handled by this lane

    // A 16x32 layout (matches B): element e of the v16h in lane L maps to
    // K = e + 16*(L>>4)
    int kmap[16];
#pragma unroll
    for (int e = 0; e < 16; ++e)
        kmap[e] = e + hi * 16;

    union AH { v16h v; _Float16 h[16]; };

    for (int tile = blockIdx.x * WAVES_PER_BLOCK + wid; tile < N_TILES;
         tile += gridDim.x * WAVES_PER_BLOCK) {
        const int row = tile * 16 + ncol;                    // lanes L and L+16: same row
        const float4* zp = (const float4*)(z + (size_t)row * F_DIM);
        float4 a4 = zp[0], b4 = zp[1], c4 = zp[2], d4 = zp[3];
        float zr[F_DIM] = {a4.x,a4.y,a4.z,a4.w, b4.x,b4.y,b4.z,b4.w,
                           c4.x,c4.y,c4.z,c4.w, d4.x,d4.y,d4.z,d4.w};
        int aIdx[F_DIM];
        for (int f = 0; f < F_DIM; ++f)
            aIdx[f] = ssorted_left(&E[f * NEDGE], zr[f]);    // == sample_bin + 1, in [0,101]

        v8f acc = {};
        for (int f = 0; f < F_DIM; ++f) {
            const int av = aIdx[f];
            const _Float16* rowB = &sT2[(f * C_CLUST + ncol) * A_PAD];
#pragma unroll
            for (int kt = 0; kt < 4; ++kt) {
                const int k0 = kt * 32;
                bool hit = (av >= k0) && (av < k0 + 32);
                if (__ballot(hit) != 0ull) {                 // wave-uniform skip, EXEC stays full
                    AH amat;
#pragma unroll
                    for (int e = 0; e < 16; ++e)
                        amat.h[e] = (av == k0 + kmap[e]) ? (_Float16)1.0f : (_Float16)0.0f;
                    // B 32x16 f16 layout: lane = column, K = e + 16*hi -> one 32B LDS chunk
                    v16h bmat = *(const v16h*)(rowB + k0 + hi * 16);
                    acc = __builtin_amdgcn_wmma_f32_16x16x32_f16(
                        false, amat.v, false, bmat, (short)0, acc, false, false);
                }
            }
        }

        // acc VGPR r holds D[m][n], m = r + 8*hi, n = ncol.
        // q = 1/(1+sqrt(mass^2)); row-normalize across the 16 lanes sharing m.
#pragma unroll
        for (int r = 0; r < 8; ++r) {
            int m = r + 8 * hi;
            float q = 1.0f / (1.0f + sqrtf(fmaxf(acc[r], 0.0f)));
            float s = q;
            s += __shfl_xor(s, 1);
            s += __shfl_xor(s, 2);
            s += __shfl_xor(s, 4);
            s += __shfl_xor(s, 8);
            out[(size_t)(tile * 16 + m) * C_CLUST + ncol] = q / s;
        }
    }
}

// -------------------------- launcher ---------------------------------------

extern "C" void kernel_launch(void* const* d_in, const int* in_sizes, int n_in,
                              void* d_out, int out_size, void* d_ws, size_t ws_size,
                              hipStream_t stream) {
    const float* z  = (const float*)d_in[0];
    const float* cl = (const float*)d_in[1];
    char* ws = (char*)d_ws;
    float*    lo  = (float*)(ws + WS_LO);
    float*    hic = (float*)(ws + WS_HI);
    unsigned* cnt = (unsigned*)(ws + WS_CNT);
    float*    edg = (float*)(ws + WS_EDG);
    _Float16* t2  = (_Float16*)(ws + WS_T2);
    float*    out = (float*)d_out;

    k_init  <<<1,   256, 0, stream>>>(lo, hic, cnt);
    k_minmax<<<400, 256, 0, stream>>>(z, cl, lo, hic);
    k_edges <<<1,   256, 0, stream>>>(lo, hic, edg);
    k_hist  <<<200, 256, 0, stream>>>(z, cl, edg, cnt);
    k_table <<<1,   256, 0, stream>>>(cl, edg, cnt, t2);
    k_mass_wmma<<<256, 256, 0, stream>>>(z, edg, t2, out);
}
